// LeNet_5_81054622810342
// MI455X (gfx1250) — matrix operand order repaired and checked by reading, not compiled, not run on hardware
//
#include <hip/hip_runtime.h>
#include <math.h>

typedef __attribute__((ext_vector_type(16))) _Float16 v16h;
typedef __attribute__((ext_vector_type(8)))  _Float16 v8h;
typedef __attribute__((ext_vector_type(8)))  float    v8f;

#define BATCH 8192

// LeNet-5 C3 sparse connectivity
__constant__ int cIDX3[6][3] = {{0,1,2},{1,2,3},{2,3,4},{3,4,5},{0,4,5},{0,1,5}};
__constant__ int cIDX4[9][4] = {{0,1,2,3},{1,2,3,4},{2,3,4,5},{0,3,4,5},{0,1,4,5},
                                {0,1,2,5},{0,1,3,4},{1,2,4,5},{0,2,3,5}};

// ---------------------------------------------------------------------------
// Kernel 1: C1 conv (1->6, 5x5, 32x32->28x28) + bias + tanh + 2x2 avgpool
// One thread per pooled output element. act1 layout: [B][6][14][14] f16.
// ---------------------------------------------------------------------------
__global__ void __launch_bounds__(256) c1_pool_kernel(
    const float* __restrict__ x, const float* __restrict__ w,
    const float* __restrict__ bias, _Float16* __restrict__ act1) {
  int idx = blockIdx.x * blockDim.x + threadIdx.x;
  if (idx >= BATCH * 6 * 14 * 14) return;
  int px = idx % 14; int t = idx / 14;
  int py = t % 14;   t /= 14;
  int c  = t % 6;    int b = t / 6;
  const float* xb = x + (size_t)b * 1024;   // 32*32
  const float* wc = w + c * 25;
  float bv = bias[c];
  float s = 0.f;
  #pragma unroll
  for (int dy = 0; dy < 2; ++dy) {
    #pragma unroll
    for (int dx = 0; dx < 2; ++dx) {
      int iy0 = 2 * py + dy, ix0 = 2 * px + dx;
      float acc = bv;
      #pragma unroll
      for (int ky = 0; ky < 5; ++ky)
        #pragma unroll
        for (int kx = 0; kx < 5; ++kx)
          acc = fmaf(xb[(iy0 + ky) * 32 + ix0 + kx], wc[ky * 5 + kx], acc);
      s += tanhf(acc);
    }
  }
  act1[idx] = (_Float16)(0.25f * s);
}

// ---------------------------------------------------------------------------
// Kernel 2: pack dense C3 weight [16][6*5*5] (scattered from sparse tables)
// into WMMA B-fragment order: wfrag3[kt][lane][e] (f16), K padded 150->160.
// B layout: n = lane%16, k = kt*32 + (e/8)*16 + (lane/16)*8 + e%8.
// ---------------------------------------------------------------------------
__global__ void prep_w3_kernel(const float* __restrict__ w3,
                               const float* __restrict__ w4,
                               const float* __restrict__ w6,
                               _Float16* __restrict__ wfrag3) {
  int idx = blockIdx.x * blockDim.x + threadIdx.x;
  if (idx >= 5 * 32 * 16) return;
  int e    = idx & 15;
  int lane = (idx >> 4) & 31;
  int kt   = idx >> 9;
  int n = lane & 15;
  int k = kt * 32 + ((e >> 3) << 4) + ((lane >> 4) << 3) + (e & 7);
  float val = 0.f;
  if (k < 150) {
    int c = k / 25, r = k % 25;
    if (n < 6) {
      #pragma unroll
      for (int j = 0; j < 3; ++j)
        if (cIDX3[n][j] == c) val = w3[(n * 3 + j) * 25 + r];
    } else if (n < 15) {
      int row = n - 6;
      #pragma unroll
      for (int j = 0; j < 4; ++j)
        if (cIDX4[row][j] == c) val = w4[(row * 4 + j) * 25 + r];
    } else {
      val = w6[c * 25 + r];
    }
  }
  wfrag3[idx] = (_Float16)val;
}

// ---------------------------------------------------------------------------
// Kernel 3: C3 implicit GEMM via WMMA, LDS-staged.
// One workgroup per image (224 threads = 7 waves). The f16 image tile
// (1176 halves + 8 zero-pad = 2368 B) is loaded into LDS once, coalesced.
// Each wave computes a 16x16 D tile: M = output position in [0,112) (100
// valid), N = 16 channels, K = 150 -> 5 k-tiles.
// A-fragment: both khalf=0/1 LDS indices are COMPILE-TIME constants
// (k->(c,ky,kx)->offset folds; k>=150 maps to the zeroed pad slot), so each
// element is one v_cndmask + one ds_load_u16 — no runtime div/mod.
// Epilogue: +bias, tanh -> conv3 [B][16][100] f16.
// ---------------------------------------------------------------------------
__global__ void __launch_bounds__(224) c3_wmma_kernel(
    const _Float16* __restrict__ act1, const _Float16* __restrict__ wfrag3,
    const float* __restrict__ b3, const float* __restrict__ b4,
    const float* __restrict__ b6, _Float16* __restrict__ conv3) {
  __shared__ _Float16 simg[1184];          // 1176 image + 8 zero pad
  const int ZSLOT = 1176;
  int b = blockIdx.x;
  const _Float16* src = act1 + (size_t)b * 1176;
  for (int i = threadIdx.x; i < 1184; i += 224)
    simg[i] = (i < 1176) ? src[i] : (_Float16)0.f;
  __syncthreads();

  int wave = threadIdx.x >> 5;    // 0..6
  int lane = threadIdx.x & 31;
  bool khalf = (lane & 16) != 0;
  int p = wave * 16 + (lane & 15);
  int pc = (p < 100) ? p : 99;    // clamp addressing for pad rows
  int ab = (pc / 10) * 14 + (pc % 10);   // oy*14 + ox, runtime per lane

  v8f acc = {};
  #pragma unroll
  for (int kt = 0; kt < 5; ++kt) {
    v16h a;
    #pragma unroll
    for (int e = 0; e < 16; ++e) {
      const int k0 = kt * 32 + ((e >> 3) << 4) + (e & 7);  // khalf=0 (const)
      const int k1 = k0 + 8;                               // khalf=1 (const)
      const int o0 = (k0 < 150)
          ? ((k0 / 25) * 196 + ((k0 % 25) / 5) * 14 + ((k0 % 25) % 5)) : -1;
      const int o1 = (k1 < 150)
          ? ((k1 / 25) * 196 + ((k1 % 25) / 5) * 14 + ((k1 % 25) % 5)) : -1;
      int idx0 = (o0 >= 0) ? (ab + o0) : ZSLOT;   // folds: add or const
      int idx1 = (o1 >= 0) ? (ab + o1) : ZSLOT;
      int idx = khalf ? idx1 : idx0;              // one v_cndmask
      a[e] = simg[idx];                           // ds_load_u16
    }
    v16h bf = *(const v16h*)(wfrag3 + ((kt * 32 + lane) << 4));
    acc = __builtin_amdgcn_wmma_f32_16x16x32_f16(
        false, a, false, bf, (short)0, acc, false, false);
  }

  int n = lane & 15;
  float bias = (n < 6) ? b3[n] : ((n < 15) ? b4[n - 6] : b6[0]);
  _Float16* dst = conv3 + ((size_t)b * 16 + n) * 100;
  #pragma unroll
  for (int v = 0; v < 8; ++v) {
    int pp = wave * 16 + v + (khalf ? 8 : 0);
    if (pp < 100) dst[pp] = (_Float16)tanhf(acc[v] + bias);
  }
}

// ---------------------------------------------------------------------------
// Kernel 4: 2x2 avgpool of conv3 [B][16][10][10] -> im2col rows for C5:
// act2 f16 [B][416]: k = c*25 + py*5 + px for k<400, zero pad k=400..415
// (so C5 needs no K tail handling).
// ---------------------------------------------------------------------------
__global__ void __launch_bounds__(256) pool_c3_kernel(
    const _Float16* __restrict__ conv3, _Float16* __restrict__ act2) {
  int idx = blockIdx.x * blockDim.x + threadIdx.x;
  if (idx >= BATCH * 416) return;
  int k = idx % 416;
  int b = idx / 416;
  if (k >= 400) { act2[idx] = (_Float16)0.f; return; }
  int c = k / 25, r = k % 25, py = r / 5, px = r % 5;
  const _Float16* s = conv3 + ((size_t)b * 16 + c) * 100;
  float sum = (float)s[(2 * py) * 10 + 2 * px]     + (float)s[(2 * py) * 10 + 2 * px + 1]
            + (float)s[(2 * py + 1) * 10 + 2 * px] + (float)s[(2 * py + 1) * 10 + 2 * px + 1];
  act2[idx] = (_Float16)(0.25f * sum);
}

// ---------------------------------------------------------------------------
// Kernel 5: pack c5_w [120][400] into B-fragments: wfrag5[nt][kt][lane][e],
// nt in 0..7 (N padded 120->128), kt in 0..12 (K padded 400->416).
// ---------------------------------------------------------------------------
__global__ void prep_w5_kernel(const float* __restrict__ w5,
                               _Float16* __restrict__ wfrag5) {
  int idx = blockIdx.x * blockDim.x + threadIdx.x;
  if (idx >= 8 * 13 * 512) return;
  int e    = idx & 15;
  int lane = (idx >> 4) & 31;
  int kt   = (idx >> 9) % 13;
  int nt   = idx / (13 * 512);
  int n = nt * 16 + (lane & 15);
  int k = kt * 32 + ((e >> 3) << 4) + ((lane >> 4) << 3) + (e & 7);
  float val = (n < 120 && k < 400) ? w5[n * 400 + k] : 0.f;
  wfrag5[idx] = (_Float16)val;
}

// ---------------------------------------------------------------------------
// Kernel 6: C5 GEMM via WMMA, 1x2 N-blocked: M = B, N = 120(->128, two
// 16-wide tiles per wave), K = 416 (pre-padded). Each wave loads one
// A-fragment per k-tile (two aligned 16-byte loads) and multiplies it into
// TWO B-fragments -> two accumulators: halves A traffic vs 1 tile/wave.
// Epilogue: +bias, tanh -> act3 [B][120] f32.
// ---------------------------------------------------------------------------
__global__ void __launch_bounds__(256) c5_wmma_kernel(
    const _Float16* __restrict__ act2, const _Float16* __restrict__ wfrag5,
    const float* __restrict__ c5_b, float* __restrict__ act3) {
  int wave = (int)((blockIdx.x * blockDim.x + threadIdx.x) >> 5);  // 2048 waves
  int lane = threadIdx.x & 31;
  int mt = wave >> 2;        // 512 m-tiles
  int np = wave & 3;         // n-pair: tiles 2np, 2np+1
  int khalf = lane >> 4;
  int m = mt * 16 + (lane & 15);
  const _Float16* arow = act2 + (size_t)m * 416;

  v8f acc0 = {}, acc1 = {};
  #pragma unroll
  for (int kt = 0; kt < 13; ++kt) {
    int base = kt * 32 + (khalf << 3);
    v8h lo = *(const v8h*)(arow + base);        // k = base .. base+7
    v8h hi = *(const v8h*)(arow + base + 16);   // k = base+16 .. base+23
    v16h a;
    #pragma unroll
    for (int e = 0; e < 8; ++e) { a[e] = lo[e]; a[e + 8] = hi[e]; }
    v16h bf0 = *(const v16h*)(wfrag5 + (((size_t)((2 * np)     * 13 + kt) * 32 + lane) << 4));
    v16h bf1 = *(const v16h*)(wfrag5 + (((size_t)((2 * np + 1) * 13 + kt) * 32 + lane) << 4));
    acc0 = __builtin_amdgcn_wmma_f32_16x16x32_f16(
        false, a, false, bf0, (short)0, acc0, false, false);
    acc1 = __builtin_amdgcn_wmma_f32_16x16x32_f16(
        false, a, false, bf1, (short)0, acc1, false, false);
  }

  #pragma unroll
  for (int half = 0; half < 2; ++half) {
    int n = (2 * np + half) * 16 + (lane & 15);
    if (n < 120) {
      float bias = c5_b[n];
      #pragma unroll
      for (int v = 0; v < 8; ++v) {
        int mm = mt * 16 + v + (khalf << 3);
        float val = (half == 0) ? acc0[v] : acc1[v];
        act3[(size_t)mm * 120 + n] = tanhf(val + bias);
      }
    }
  }
}

// ---------------------------------------------------------------------------
// Kernel 7: F6: act4 = tanh(act3 @ l1_w.T + l1_b), [B][84]
// ---------------------------------------------------------------------------
__global__ void __launch_bounds__(256) fc84_kernel(
    const float* __restrict__ act3, const float* __restrict__ w,
    const float* __restrict__ bias, float* __restrict__ act4) {
  int idx = blockIdx.x * blockDim.x + threadIdx.x;
  if (idx >= BATCH * 84) return;
  int j = idx % 84; int b = idx / 84;
  const float* a  = act3 + (size_t)b * 120;
  const float* wr = w + j * 120;
  float s = bias[j];
  #pragma unroll 8
  for (int k = 0; k < 120; ++k) s = fmaf(a[k], wr[k], s);
  act4[idx] = tanhf(s);
}

// ---------------------------------------------------------------------------
// Kernel 8: output layer: out = act4 @ l2_w.T + l2_b, [B][10]
// ---------------------------------------------------------------------------
__global__ void __launch_bounds__(256) fc10_kernel(
    const float* __restrict__ act4, const float* __restrict__ w,
    const float* __restrict__ bias, float* __restrict__ out) {
  int idx = blockIdx.x * blockDim.x + threadIdx.x;
  if (idx >= BATCH * 10) return;
  int o = idx % 10; int b = idx / 10;
  const float* a  = act4 + (size_t)b * 84;
  const float* wr = w + o * 84;
  float s = bias[o];
  #pragma unroll 4
  for (int k = 0; k < 84; ++k) s = fmaf(a[k], wr[k], s);
  out[idx] = s;
}

// ---------------------------------------------------------------------------
// Workspace layout (bytes, all 256-aligned):
//   act1   f16 [B][6][14][14]   19,267,584   @ 0
//   conv3  f16 [B][16][100]     26,214,400   @ 19,267,584
//   act2   f16 [B][416]          6,815,744   @ 45,481,984
//   act3   f32 [B][120]          3,932,160   @ 52,297,728
//   act4   f32 [B][84]           2,752,512   @ 56,229,888
//   wfrag3 f16 [5][32][16]           5,120   @ 58,982,400
//   wfrag5 f16 [8][13][32][16]     106,496   @ 58,987,520
//   total: 59,094,016
// ---------------------------------------------------------------------------
extern "C" void kernel_launch(void* const* d_in, const int* in_sizes, int n_in,
                              void* d_out, int out_size, void* d_ws, size_t ws_size,
                              hipStream_t stream) {
  (void)in_sizes; (void)n_in; (void)out_size; (void)ws_size;

  const float* x     = (const float*)d_in[0];
  const float* c1_w  = (const float*)d_in[1];
  const float* c1_b  = (const float*)d_in[2];
  const float* c3_w3 = (const float*)d_in[3];
  const float* c3_b3 = (const float*)d_in[4];
  const float* c3_w4 = (const float*)d_in[5];
  const float* c3_b4 = (const float*)d_in[6];
  const float* c3_w6 = (const float*)d_in[7];
  const float* c3_b6 = (const float*)d_in[8];
  const float* c5_w  = (const float*)d_in[9];
  const float* c5_b  = (const float*)d_in[10];
  const float* l1_w  = (const float*)d_in[11];
  const float* l1_b  = (const float*)d_in[12];
  const float* l2_w  = (const float*)d_in[13];
  const float* l2_b  = (const float*)d_in[14];
  float* out = (float*)d_out;

  char* base = (char*)d_ws;
  _Float16* act1   = (_Float16*)(base + 0);
  _Float16* conv3  = (_Float16*)(base + 19267584);
  _Float16* act2   = (_Float16*)(base + 45481984);
  float*    act3   = (float*)   (base + 52297728);
  float*    act4   = (float*)   (base + 56229888);
  _Float16* wfrag3 = (_Float16*)(base + 58982400);
  _Float16* wfrag5 = (_Float16*)(base + 58987520);

  // Weight packing (tiny)
  prep_w3_kernel<<<10, 256, 0, stream>>>(c3_w3, c3_w4, c3_w6, wfrag3);
  prep_w5_kernel<<<208, 256, 0, stream>>>(c5_w, wfrag5);

  // C1 + tanh + pool: B*6*14*14 = 9,633,792 threads
  c1_pool_kernel<<<(BATCH * 6 * 14 * 14 + 255) / 256, 256, 0, stream>>>(
      x, c1_w, c1_b, act1);

  // C3 WMMA, LDS-staged: one workgroup (7 waves) per image
  c3_wmma_kernel<<<BATCH, 224, 0, stream>>>(act1, wfrag3, c3_b3, c3_b4, c3_b6, conv3);

  // Pool -> padded f16 im2col rows: B*416 threads
  pool_c3_kernel<<<(BATCH * 416 + 255) / 256, 256, 0, stream>>>(conv3, act2);

  // C5 WMMA, 1x2 N-blocked: 2048 waves -> 256 blocks
  c5_wmma_kernel<<<256, 256, 0, stream>>>(act2, wfrag5, c5_b, act3);

  // F6 and output layer
  fc84_kernel<<<(BATCH * 84 + 255) / 256, 256, 0, stream>>>(act3, l1_w, l1_b, act4);
  fc10_kernel<<<(BATCH * 10 + 255) / 256, 256, 0, stream>>>(act4, l2_w, l2_b, out);
}
